// MacDeformableCrossAttention_71983651881166
// MI455X (gfx1250) — compile-verified, matrix operand order checked by reading, not run on hardware
//
#include <hip/hip_runtime.h>
#include <hip/hip_bf16.h>

// ---------------- problem constants ----------------
#define Bc   4
#define NQ   1000
#define Cc   256
#define Hc   200
#define Wc   100
#define HWc  20000          // Hc*Wc
#define NHc  8
#define NPc  4
#define HDc  32

typedef __attribute__((ext_vector_type(16))) _Float16 v16h;
typedef __attribute__((ext_vector_type(8)))  _Float16 v8h;
typedef __attribute__((ext_vector_type(8)))  float    v8f;

// ---------------- workspace layout (bytes) ----------------
#define WS_WVT   0            // f16 [256][256]  (131072)
#define WS_WOUTT 131072       // f16 [256][256]  (131072)
#define WS_VMT   262144       // f16 [B*HW][256] (40,960,000)
#define WS_V     41222144     // f16 [B*HW][256] (40,960,000)
#define WS_PX    82182144     // f32 [4000*32]   (512,000)
#define WS_PY    82694144     // f32 [4000*32]   (512,000)
#define WS_ATT   83206144     // f32 [4000*32]   (512,000)
#define WS_AGG   83718144     // f16 [4000][256] (2,048,000)  total ~85.8 MB

// ---------------- kernel 1: weight convert + transpose ----------------
__global__ void k_cvt_weights(const float* __restrict__ Wv,
                              const float* __restrict__ Wout,
                              _Float16* __restrict__ WvT,
                              _Float16* __restrict__ WoutT) {
    int idx = blockIdx.x * blockDim.x + threadIdx.x;   // 0..65535
    int k = idx >> 8, n = idx & 255;
    WvT  [n * 256 + k] = (_Float16)Wv  [idx];
    WoutT[n * 256 + k] = (_Float16)Wout[idx];
}

// ---------------- kernel 2: LDS-tiled transpose ------------------------
// value_map f32 [B][C][HW] -> vmT f16 [B][HW][C], 32x32 tiles.
// Reads coalesced over hw, writes coalesced over c (64B per wave).
__global__ void k_transpose_vm(const float* __restrict__ vm,
                               _Float16* __restrict__ vmT) {
    __shared__ _Float16 tile[32][33];       // pad -> conflict-free columns

    int id  = blockIdx.x;                    // 0 .. B*8*625-1
    int b   = id / 5000;
    int rem = id - b * 5000;
    int ct  = rem / 625;                     // C tile   (8)
    int ht  = rem - ct * 625;                // HW tile  (625)
    int c0  = ct << 5;
    int hw0 = ht << 5;

    int tx = threadIdx.x & 31;
    int ty = threadIdx.x >> 5;               // 0..7

#pragma unroll
    for (int i = 0; i < 4; ++i) {
        int c = c0 + ty + i * 8;
        tile[ty + i * 8][tx] =
            (_Float16)vm[((size_t)b * Cc + c) * HWc + hw0 + tx];
    }
    __syncthreads();
#pragma unroll
    for (int i = 0; i < 4; ++i) {
        int r = ty + i * 8;                  // row within hw tile
        vmT[((size_t)b * HWc + hw0 + r) * Cc + c0 + tx] = tile[tx][r];
    }
}

// ---------------- WMMA 16x16x32 f16 fragment load ----------------------
// 16-bit A/B layout (CDNA5 ISA 7.12.2): lanes 0-15 -> row=lane, K chunks
// {0..7, 16..23}; lanes 16-31 -> row=lane-16, K chunks {8..15, 24..31}.
__device__ __forceinline__ v16h load_frag(const _Float16* __restrict__ base,
                                          int ld, int lane) {
    int r  = lane & 15;
    int k0 = (lane >> 4) << 3;               // 0 or 8
    const _Float16* p = base + r * ld + k0;
    v8h lo = *(const v8h*)(p);               // K = k0 .. k0+7   (16B)
    v8h hi = *(const v8h*)(p + 16);          // K = k0+16..k0+23 (16B)
    v16h f;
#pragma unroll
    for (int i = 0; i < 8; ++i) { f[i] = lo[i]; f[i + 8] = hi[i]; }
    return f;
}

// ---------------- M x 256 * 256 x 256 WMMA GEMM, 16x64 per wave --------
// A: f16 [M][256] row-major, Bt: f16 [256][256] = B^T (N-major).
// Each wave owns one 16x64 output strip: A fragment reused across 4 N-tiles
// (4x less A traffic), 4 independent WMMA chains hide XDL latency.
template <bool HALF_OUT>
__global__ void k_gemm256(const _Float16* __restrict__ A,
                          const _Float16* __restrict__ Bt,
                          const float* __restrict__ bias,
                          float* __restrict__ outF,
                          _Float16* __restrict__ outH) {
    int wave = (blockIdx.x * blockDim.x + threadIdx.x) >> 5;
    int lane = threadIdx.x & 31;
    int tM = wave >> 2;                       // M tile
    int gN = wave & 3;                        // N group: cols gN*64 .. +63

    const _Float16* a0 = A  + (size_t)tM * 16 * 256;
    const _Float16* b0 = Bt + (size_t)gN * 64 * 256;

    v8f acc0 = {}, acc1 = {}, acc2 = {}, acc3 = {};
#pragma unroll
    for (int kk = 0; kk < 256; kk += 32) {
        v16h af  = load_frag(a0 + kk,            256, lane);
        v16h bf0 = load_frag(b0 + kk,            256, lane);
        v16h bf1 = load_frag(b0 + 16 * 256 + kk, 256, lane);
        v16h bf2 = load_frag(b0 + 32 * 256 + kk, 256, lane);
        v16h bf3 = load_frag(b0 + 48 * 256 + kk, 256, lane);
        acc0 = __builtin_amdgcn_wmma_f32_16x16x32_f16(false, af, false, bf0,
                                                      (short)0, acc0, false, false);
        acc1 = __builtin_amdgcn_wmma_f32_16x16x32_f16(false, af, false, bf1,
                                                      (short)0, acc1, false, false);
        acc2 = __builtin_amdgcn_wmma_f32_16x16x32_f16(false, af, false, bf2,
                                                      (short)0, acc2, false, false);
        acc3 = __builtin_amdgcn_wmma_f32_16x16x32_f16(false, af, false, bf3,
                                                      (short)0, acc3, false, false);
    }

    // C/D layout: lane -> N = lane&15, M-half = lane>>4, element i -> row m0+i
    int nl = lane & 15;
    int m0 = tM * 16 + ((lane >> 4) << 3);
    v8f accs[4] = {acc0, acc1, acc2, acc3};
#pragma unroll
    for (int j = 0; j < 4; ++j) {
        int n  = gN * 64 + j * 16 + nl;
        float bn = bias[n];
#pragma unroll
        for (int i = 0; i < 8; ++i) {
            float r = accs[j][i] + bn;
            if (HALF_OUT) outH[(size_t)(m0 + i) * 256 + n] = (_Float16)r;
            else          outF[(size_t)(m0 + i) * 256 + n] = r;
        }
    }
}

// ---------------- kernel 4: offsets + attention softmax ----------------
// one block (128 threads) per (b,q)
__global__ void k_offatt(const float* __restrict__ query,
                         const float* __restrict__ ref,
                         const float* __restrict__ Woff,
                         const float* __restrict__ boff,
                         const float* __restrict__ Watt,
                         const float* __restrict__ batt,
                         float* __restrict__ px,
                         float* __restrict__ py,
                         float* __restrict__ attw) {
    int bq  = blockIdx.x;
    int tid = threadIdx.x;
    __shared__ float q[256];
    __shared__ float latt[32];

    const float* qr = query + (size_t)bq * 256;
    q[tid]       = qr[tid];
    q[tid + 128] = qr[tid + 128];
    __syncthreads();

    if (tid < 64) {
        float acc = boff[tid];
#pragma unroll 8
        for (int k = 0; k < 256; ++k) acc += q[k] * Woff[k * 64 + tid];
        int   xy  = tid & 1;
        float r   = ref[(size_t)bq * 2 + xy];
        float loc = r + acc;
        // align_corners=False: x uses W=100, y uses H=200
        float pix = xy == 0 ? loc * (float)Wc - 0.5f : loc * (float)Hc - 0.5f;
        int   o   = bq * 32 + (tid >> 1);     // [b,q,nh,p] flat
        if (xy == 0) px[o] = pix; else py[o] = pix;
    } else if (tid < 96) {
        int j = tid - 64;
        float acc = batt[j];
#pragma unroll 8
        for (int k = 0; k < 256; ++k) acc += q[k] * Watt[k * 32 + j];
        latt[j] = acc;
    }
    __syncthreads();

    if (tid >= 64 && tid < 96) {
        int j = tid - 64;
        int g = j & ~3;                       // softmax over NP=4 per head
        float m = fmaxf(fmaxf(latt[g], latt[g + 1]), fmaxf(latt[g + 2], latt[g + 3]));
        float e0 = __expf(latt[g    ] - m);
        float e1 = __expf(latt[g + 1] - m);
        float e2 = __expf(latt[g + 2] - m);
        float e3 = __expf(latt[g + 3] - m);
        attw[bq * 32 + j] = __expf(latt[j] - m) / (e0 + e1 + e2 + e3);
    }
}

// ---------------- kernel 5: bilinear sampling + aggregation ------------
// one wave32 per (b,q,nh); lane = channel-in-head; wave reads 64B slices
__global__ void k_sample(const _Float16* __restrict__ v,
                         const float* __restrict__ px,
                         const float* __restrict__ py,
                         const float* __restrict__ attw,
                         _Float16* __restrict__ agg) {
    int gtid = blockIdx.x * blockDim.x + threadIdx.x;
    int wave = gtid >> 5;
    int lane = gtid & 31;
    int nh = wave & (NHc - 1);
    int bq = wave >> 3;
    int b  = bq / NQ;

    const _Float16* vb = v + (size_t)b * HWc * Cc + nh * HDc + lane;
    int base = bq * 32 + nh * 4;

    float acc = 0.f;
#pragma unroll
    for (int p = 0; p < NPc; ++p) {
        float fx = px[base + p];
        float fy = py[base + p];
        float aw = attw[base + p];
        float x0f = floorf(fx), y0f = floorf(fy);
        int   x0 = (int)x0f,   y0 = (int)y0f;
        float wx = fx - x0f,   wy = fy - y0f;

        float s = 0.f;
#pragma unroll
        for (int cy = 0; cy < 2; ++cy) {
#pragma unroll
            for (int cx = 0; cx < 2; ++cx) {
                int xx = x0 + cx, yy = y0 + cy;
                float w = (cx ? wx : 1.f - wx) * (cy ? wy : 1.f - wy);
                if (xx >= 0 && xx < Wc && yy >= 0 && yy < Hc) {
                    float g = (float)vb[(size_t)(yy * Wc + xx) * Cc];
                    s += w * g;
                }
            }
        }
        acc += aw * s;
    }
    agg[(size_t)bq * Cc + nh * HDc + lane] = (_Float16)acc;
}

// ---------------- launcher ----------------
extern "C" void kernel_launch(void* const* d_in, const int* in_sizes, int n_in,
                              void* d_out, int out_size, void* d_ws, size_t ws_size,
                              hipStream_t stream) {
    const float* query = (const float*)d_in[0];   // [4,1000,256]
    const float* refp  = (const float*)d_in[1];   // [4,1000,2]
    const float* vmap  = (const float*)d_in[2];   // [4,256,200,100]
    const float* Wv    = (const float*)d_in[3];   // [256,256]
    const float* bv    = (const float*)d_in[4];   // [256]
    const float* Woff  = (const float*)d_in[5];   // [256,64]
    const float* boff  = (const float*)d_in[6];   // [64]
    const float* Watt  = (const float*)d_in[7];   // [256,32]
    const float* batt  = (const float*)d_in[8];   // [32]
    const float* Wout  = (const float*)d_in[9];   // [256,256]
    const float* bout  = (const float*)d_in[10];  // [256]
    float* out = (float*)d_out;                   // [4,1000,256] f32

    char* ws = (char*)d_ws;
    _Float16* WvT   = (_Float16*)(ws + WS_WVT);
    _Float16* WoutT = (_Float16*)(ws + WS_WOUTT);
    _Float16* vmT   = (_Float16*)(ws + WS_VMT);
    _Float16* vproj = (_Float16*)(ws + WS_V);
    float*    px    = (float*)   (ws + WS_PX);
    float*    py    = (float*)   (ws + WS_PY);
    float*    attw  = (float*)   (ws + WS_ATT);
    _Float16* agg   = (_Float16*)(ws + WS_AGG);

    // 1) weights -> f16, transposed (B-matrix N-major)
    k_cvt_weights<<<256, 256, 0, stream>>>(Wv, Wout, WvT, WoutT);

    // 2) value_map [B,C,HW] f32 -> [B,HW,C] f16, LDS-tiled (coalesced both ways)
    k_transpose_vm<<<Bc * 8 * 625, 256, 0, stream>>>(vmap, vmT);

    // 3) value projection GEMM: [80000 x 256] x [256 x 256] -> f16 v
    //    16x64 per wave: 5000 M-tiles * 4 N-groups = 20000 waves, 8/block
    k_gemm256<true><<<2500, 256, 0, stream>>>(vmT, WvT, bv, nullptr, vproj);

    // 4) offsets + attention softmax (one block per (b,q))
    k_offatt<<<Bc * NQ, 128, 0, stream>>>(query, refp, Woff, boff, Watt, batt,
                                          px, py, attw);

    // 5) bilinear sampling + aggregation -> agg f16 [4000][256]
    k_sample<<<4000, 256, 0, stream>>>(vproj, px, py, attw, agg);

    // 6) output GEMM: [4000 x 256] x [256 x 256] + bout -> f32 out
    //    250 M-tiles * 4 N-groups = 1000 waves, 8/block
    k_gemm256<false><<<125, 256, 0, stream>>>(agg, WoutT, bout, out, nullptr);
}
